// PermProjHead_59957743452747
// MI455X (gfx1250) — matrix operand order repaired and checked by reading, not compiled, not run on hardware
//
#include <hip/hip_runtime.h>
#include <math.h>

// ---------------------------------------------------------------------------
// Problem constants
// ---------------------------------------------------------------------------
#define NB   32          // batch
#define TT   512         // tokens
#define DD   1024        // feature dim
#define CC   120         // classes
#define NBR  3           // branches (anc/pos/neg)
#define INV_SCALE (1.0f/64.0f)   // 1/(2*sqrt(1024))

typedef __attribute__((ext_vector_type(16))) __bf16 v16bf;
typedef __attribute__((ext_vector_type(8)))  __bf16 v8bf;
typedef __attribute__((ext_vector_type(8)))  float  v8f;

static __device__ __forceinline__ v16bf combine16(v8bf lo, v8bf hi) {
    return __builtin_shufflevector(lo, hi, 0,1,2,3,4,5,6,7,8,9,10,11,12,13,14,15);
}
static __device__ __forceinline__ v8f wmma_bf16(v16bf a, v16bf b, v8f c) {
    return __builtin_amdgcn_wmma_f32_16x16x32_bf16(
        /*neg_a=*/false, a, /*neg_b=*/false, b,
        /*c_mod=*/(short)0, c, /*reuse_a=*/false, /*reuse_b=*/false);
}

// ---------------------------------------------------------------------------
// Kernel 1: W1 (D x D, fp32, [k][n]) -> Wt (bf16, transposed [n][k])
// ---------------------------------------------------------------------------
__global__ void k_w1_transpose(const float* __restrict__ W1, __bf16* __restrict__ Wt) {
    __shared__ float tile[32][33];
    int bx = blockIdx.x;            // k-tile
    int by = blockIdx.y;            // n-tile
    int tx = threadIdx.x & 31;
    int ty = threadIdx.x >> 5;      // 0..7
    for (int k = 0; k < 32; k += 8)
        tile[ty + k][tx] = W1[(size_t)(bx*32 + ty + k)*DD + by*32 + tx];
    __syncthreads();
    for (int k = 0; k < 32; k += 8)
        Wt[(size_t)(by*32 + ty + k)*DD + bx*32 + tx] = (__bf16)tile[tx][ty + k];
}

// ---------------------------------------------------------------------------
// Kernel 2: masked row-sum of x over T + valid count.   grid=(NB, NBR)
// ---------------------------------------------------------------------------
__global__ void k_xsum(const float* __restrict__ x0, const float* __restrict__ x1,
                       const float* __restrict__ x2,
                       const int* __restrict__ m0, const int* __restrict__ m1,
                       const int* __restrict__ m2,
                       float* __restrict__ xsum, float* __restrict__ nvalid) {
    int b  = blockIdx.x, br = blockIdx.y;
    const float* x   = (br == 0) ? x0 : (br == 1) ? x1 : x2;
    const int*   msk = (br == 0) ? m0 : (br == 1) ? m1 : m2;
    int tid = threadIdx.x, lane = tid & 31, wave = tid >> 5;

    __shared__ unsigned char valid[TT];
    __shared__ int redc[8];
    int local = 0;
    for (int t = tid; t < TT; t += 256) {
        int v = (msk[b*TT + t] == 0);
        valid[t] = (unsigned char)v;
        local += v;
    }
    for (int off = 16; off; off >>= 1) local += __shfl_xor(local, off);
    if (lane == 0) redc[wave] = local;
    __syncthreads();
    if (tid == 0) {
        int s = 0;
        for (int i = 0; i < 8; ++i) s += redc[i];
        nvalid[br*NB + b] = (float)s;
    }
    for (int d = tid; d < DD; d += 256) {
        float acc = 0.f;
        const float* xb = x + (size_t)b*TT*DD + d;
        for (int t = 0; t < TT; ++t)
            if (valid[t]) acc += xb[(size_t)t*DD];
        xsum[((size_t)br*NB + b)*DD + d] = acc;
    }
}

// ---------------------------------------------------------------------------
// Kernel 3: v = xsum@W_attn + nv*b_attn ; u = W_attn @ v ; cb = b_attn . v
// grid=(NB, NBR)
// ---------------------------------------------------------------------------
__global__ void k_uv(const float* __restrict__ xsum, const float* __restrict__ nvalid,
                     const float* __restrict__ W_attn, const float* __restrict__ b_attn,
                     float* __restrict__ u, float* __restrict__ cb) {
    int b = blockIdx.x, br = blockIdx.y;
    int tid = threadIdx.x, lane = tid & 31, wave = tid >> 5;
    __shared__ float xs[DD];
    __shared__ float vs[DD];
    __shared__ float red[8];
    size_t idx = (size_t)br*NB + b;
    for (int i = tid; i < DD; i += 256) xs[i] = xsum[idx*DD + i];
    float nv = nvalid[idx];
    __syncthreads();
    for (int o = tid; o < DD; o += 256) {
        float acc = nv * b_attn[o];
        for (int i = 0; i < DD; ++i) acc += xs[i] * W_attn[(size_t)i*DD + o];
        vs[o] = acc;
    }
    __syncthreads();
    float p = 0.f;
    for (int o = tid; o < DD; o += 256) p += b_attn[o] * vs[o];
    for (int off = 16; off; off >>= 1) p += __shfl_xor(p, off);
    if (lane == 0) red[wave] = p;
    __syncthreads();
    if (tid == 0) {
        float s = 0.f;
        for (int i = 0; i < 8; ++i) s += red[i];
        cb[idx] = s;
    }
    for (int i = tid; i < DD; i += 256) {
        float acc = 0.f;
        const float* wr = W_attn + (size_t)i*DD;
        for (int o = 0; o < DD; ++o) acc += wr[o] * vs[o];
        u[idx*DD + i] = acc;
    }
}

// ---------------------------------------------------------------------------
// Kernel 4: s[t] = (x[t].u + cb)/SCALE (masked -> -inf), softmax over t.
// grid=(NB, NBR)
// ---------------------------------------------------------------------------
__global__ void k_attn(const float* __restrict__ x0, const float* __restrict__ x1,
                       const float* __restrict__ x2,
                       const int* __restrict__ m0, const int* __restrict__ m1,
                       const int* __restrict__ m2,
                       const float* __restrict__ u, const float* __restrict__ cb,
                       float* __restrict__ attn) {
    int b = blockIdx.x, br = blockIdx.y;
    const float* x   = (br == 0) ? x0 : (br == 1) ? x1 : x2;
    const int*   msk = (br == 0) ? m0 : (br == 1) ? m1 : m2;
    int tid = threadIdx.x, lane = tid & 31, wave = tid >> 5;
    size_t idx = (size_t)br*NB + b;

    __shared__ float us[DD];
    __shared__ float ss[TT];
    __shared__ float red[8];
    for (int i = tid; i < DD; i += 256) us[i] = u[idx*DD + i];
    float c = cb[idx];
    __syncthreads();

    for (int t = wave; t < TT; t += 8) {
        const float* xr = x + ((size_t)b*TT + t)*DD;
        float acc = 0.f;
        for (int d = lane; d < DD; d += 32) acc += xr[d] * us[d];
        for (int off = 16; off; off >>= 1) acc += __shfl_xor(acc, off);
        if (lane == 0)
            ss[t] = (msk[b*TT + t] != 0) ? -3.0e38f : (acc + c) * INV_SCALE;
    }
    __syncthreads();

    float m = -3.0e38f;
    for (int t = tid; t < TT; t += 256) m = fmaxf(m, ss[t]);
    for (int off = 16; off; off >>= 1) m = fmaxf(m, __shfl_xor(m, off));
    if (lane == 0) red[wave] = m;
    __syncthreads();
    if (tid == 0) {
        float mm = red[0];
        for (int i = 1; i < 8; ++i) mm = fmaxf(mm, red[i]);
        red[0] = mm;
    }
    __syncthreads();
    m = red[0];
    __syncthreads();
    float sum = 0.f;
    for (int t = tid; t < TT; t += 256) {
        float e = __expf(ss[t] - m);
        ss[t] = e;
        sum += e;
    }
    for (int off = 16; off; off >>= 1) sum += __shfl_xor(sum, off);
    if (lane == 0) red[wave] = sum;
    __syncthreads();
    if (tid == 0) {
        float s = 0.f;
        for (int i = 0; i < 8; ++i) s += red[i];
        red[0] = s;
    }
    __syncthreads();
    float inv = 1.0f / red[0];
    for (int t = tid; t < TT; t += 256) attn[idx*TT + t] = ss[t] * inv;
}

// ---------------------------------------------------------------------------
// Kernel 5 (WMMA, double-buffered): pooledh[b,:] += sum_t attn[t]*relu(x[t]@W1+b1)
// grid = (M/128, D/128, NBR); 128x128 tile per WG, 8 waves, 32x64 per wave,
// K_TILE=32, software-pipelined: global loads for step k+1 issue before the
// WMMA block of step k; convert/ds_store after WMMAs (waits overlap compute).
// ---------------------------------------------------------------------------
__global__ __launch_bounds__(256)
void k_gemm_pool(const float* __restrict__ x0, const float* __restrict__ x1,
                 const float* __restrict__ x2,
                 const __bf16* __restrict__ Wt, const float* __restrict__ b1,
                 const float* __restrict__ attn, float* __restrict__ pooledh) {
    int mblk = blockIdx.x, nblk = blockIdx.y, br = blockIdx.z;
    const float* x = (br == 0) ? x0 : (br == 1) ? x1 : x2;
    int tid = threadIdx.x, lane = tid & 31, wave = tid >> 5;
    int wm = wave & 3, wn = wave >> 2;      // 4 waves in M, 2 in N
    int m0 = mblk * 128;
    int batch = m0 >> 9;                    // 512 rows per batch, 128 | 512

    __shared__ __bf16 Xs[2][128][40];       // 32 K-cols + pad, 80B row stride
    __shared__ __bf16 Wsh[2][128][40];
    __shared__ float  pool_s[128];
    __shared__ float  attn_s[128];

    if (tid < 128) {
        pool_s[tid] = 0.f;
        attn_s[tid] = attn[((size_t)br*NB + batch)*TT + (m0 & (TT-1)) + tid];
    }

    v8f acc[2][4] = {};

    const int crx = tid & 7,  rrx = tid >> 3;   // X: 8 thr x float4 = 32 cols, 32 rows/pass
    const int cww = tid & 3,  rww = tid >> 2;   // W: 4 thr x v8bf  = 32 cols, 64 rows/pass

    const float*  xbase = x  + (size_t)m0*DD;
    const __bf16* wbase = Wt + (size_t)(nblk*128)*DD;

    float4 xreg[4];
    v8bf   wreg[2];

    auto issue_loads = [&](int k0) {
        #pragma unroll
        for (int p = 0; p < 4; ++p)
            xreg[p] = *(const float4*)&xbase[(size_t)(p*32 + rrx)*DD + k0 + crx*4];
        #pragma unroll
        for (int p = 0; p < 2; ++p)
            wreg[p] = *(const v8bf*)&wbase[(size_t)(p*64 + rww)*DD + k0 + cww*8];
    };
    auto store_tiles = [&](int buf) {
        #pragma unroll
        for (int p = 0; p < 4; ++p) {
            __bf16* d = &Xs[buf][p*32 + rrx][crx*4];
            d[0] = (__bf16)xreg[p].x; d[1] = (__bf16)xreg[p].y;
            d[2] = (__bf16)xreg[p].z; d[3] = (__bf16)xreg[p].w;
        }
        #pragma unroll
        for (int p = 0; p < 2; ++p)
            *(v8bf*)&Wsh[buf][p*64 + rww][cww*8] = wreg[p];
    };
    auto compute = [&](int buf) {
        int hiA = (lane < 16) ? 0 : 8;
        int hiB = (lane < 16) ? 0 : 8;
        v16bf A[2];
        #pragma unroll
        for (int mt = 0; mt < 2; ++mt) {
            int r = wm*32 + mt*16 + (lane & 15);
            v8bf lo = *(const v8bf*)&Xs[buf][r][hiA];
            v8bf hi = *(const v8bf*)&Xs[buf][r][16 + hiA];
            A[mt] = combine16(lo, hi);
        }
        #pragma unroll
        for (int nt = 0; nt < 4; ++nt) {
            int n = wn*64 + nt*16 + (lane & 15);
            v8bf lo = *(const v8bf*)&Wsh[buf][n][hiB];
            v8bf hi = *(const v8bf*)&Wsh[buf][n][hiB + 16];
            v16bf B = combine16(lo, hi);
            acc[0][nt] = wmma_bf16(A[0], B, acc[0][nt]);
            acc[1][nt] = wmma_bf16(A[1], B, acc[1][nt]);
        }
    };

    issue_loads(0);
    store_tiles(0);
    for (int ks = 0; ks < 32; ++ks) {
        __syncthreads();                    // buf(ks&1) staged; buf^1 reads done
        int buf = ks & 1;
        if (ks + 1 < 32) issue_loads((ks + 1) * 32);   // global loads in flight
        compute(buf);                                   // 8 WMMAs hide latency
        if (ks + 1 < 32) store_tiles(buf ^ 1);          // wait+convert+ds_store
    }
    __syncthreads();

    // finalize: bias + relu + attn weight, reduce over rows
    #pragma unroll
    for (int mt = 0; mt < 2; ++mt) {
        int rloc = wm*32 + mt*16 + ((lane < 16) ? 0 : 8);
        float av[8];
        #pragma unroll
        for (int j = 0; j < 8; ++j) av[j] = attn_s[rloc + j];
        #pragma unroll
        for (int nt = 0; nt < 4; ++nt) {
            int nloc = wn*64 + nt*16 + (lane & 15);
            float bias = b1[nblk*128 + nloc];
            float partial = 0.f;
            #pragma unroll
            for (int j = 0; j < 8; ++j) {
                float h = acc[mt][nt][j] + bias;
                h = h > 0.f ? h : 0.f;
                partial += av[j] * h;
            }
            partial += __shfl_xor(partial, 16);   // fold rows 8..15 half
            if (lane < 16) atomicAdd(&pool_s[nloc], partial);
        }
    }
    __syncthreads();
    if (tid < 128)
        atomicAdd(&pooledh[((size_t)br*NB + batch)*DD + nblk*128 + tid], pool_s[tid]);
}

// ---------------------------------------------------------------------------
// Kernel 6: pooled = pooledh@W2 + b2; layernorm; relu; @W_last + b_last
// grid=(NB, NBR)
// ---------------------------------------------------------------------------
__global__ void k_final(const float* __restrict__ pooledh, const float* __restrict__ W2,
                        const float* __restrict__ b2, const float* __restrict__ ln_g,
                        const float* __restrict__ ln_b, const float* __restrict__ W_last,
                        const float* __restrict__ b_last, float* __restrict__ out) {
    int b = blockIdx.x, br = blockIdx.y;
    int tid = threadIdx.x, lane = tid & 31, wave = tid >> 5;
    size_t idx = (size_t)br*NB + b;
    __shared__ float ph[DD];
    __shared__ float ps[DD];
    __shared__ float redA[8], redB[8];
    for (int i = tid; i < DD; i += 256) ph[i] = pooledh[idx*DD + i];
    __syncthreads();
    for (int o = tid; o < DD; o += 256) {
        float acc = b2[o];
        for (int i = 0; i < DD; ++i) acc += ph[i] * W2[(size_t)i*DD + o];
        ps[o] = acc;
    }
    __syncthreads();
    float s = 0.f, sq = 0.f;
    for (int o = tid; o < DD; o += 256) { float v = ps[o]; s += v; sq += v*v; }
    for (int off = 16; off; off >>= 1) { s += __shfl_xor(s, off); sq += __shfl_xor(sq, off); }
    if (lane == 0) { redA[wave] = s; redB[wave] = sq; }
    __syncthreads();
    if (tid == 0) {
        float S = 0.f, Q = 0.f;
        for (int i = 0; i < 8; ++i) { S += redA[i]; Q += redB[i]; }
        redA[0] = S; redB[0] = Q;
    }
    __syncthreads();
    float mu  = redA[0] * (1.0f/DD);
    float var = redB[0] * (1.0f/DD) - mu*mu;
    float inv = rsqrtf(var + 1e-12f);
    __syncthreads();
    for (int o = tid; o < DD; o += 256) {
        float v = (ps[o] - mu) * inv * ln_g[o] + ln_b[o];
        ps[o] = v > 0.f ? v : 0.f;
    }
    __syncthreads();
    if (tid < CC) {
        float acc = b_last[tid];
        for (int d = 0; d < DD; ++d) acc += ps[d] * W_last[(size_t)d*CC + tid];
        out[idx*CC + tid] = acc;
    }
}

// ---------------------------------------------------------------------------
// Host launcher
// ---------------------------------------------------------------------------
extern "C" void kernel_launch(void* const* d_in, const int* in_sizes, int n_in,
                              void* d_out, int out_size, void* d_ws, size_t ws_size,
                              hipStream_t stream) {
    const float* x0 = (const float*)d_in[0];
    const float* x1 = (const float*)d_in[1];
    const float* x2 = (const float*)d_in[2];
    const int*   m0 = (const int*)d_in[3];
    const int*   m1 = (const int*)d_in[4];
    const int*   m2 = (const int*)d_in[5];
    const float* W_attn = (const float*)d_in[6];
    const float* b_attn = (const float*)d_in[7];
    const float* W1     = (const float*)d_in[8];
    const float* b1     = (const float*)d_in[9];
    const float* W2     = (const float*)d_in[10];
    const float* b2     = (const float*)d_in[11];
    const float* ln_g   = (const float*)d_in[12];
    const float* ln_b   = (const float*)d_in[13];
    const float* W_last = (const float*)d_in[14];
    const float* b_last = (const float*)d_in[15];
    float* out = (float*)d_out;

    // workspace carve (floats), Wt 16B-aligned at end
    float* wsf     = (float*)d_ws;
    float* xsum    = wsf;                               // 3*32*1024
    float* u       = xsum + NBR*NB*DD;                  // 3*32*1024
    float* cb      = u + NBR*NB*DD;                     // 96
    float* nvalid  = cb + NBR*NB;                       // 96
    float* attn    = nvalid + NBR*NB;                   // 3*32*512
    float* pooledh = attn + NBR*NB*TT;                  // 3*32*1024
    __bf16* Wt     = (__bf16*)(pooledh + NBR*NB*DD);    // 1024*1024 bf16

    hipMemsetAsync(pooledh, 0, (size_t)NBR*NB*DD*sizeof(float), stream);

    k_w1_transpose<<<dim3(DD/32, DD/32), 256, 0, stream>>>(W1, Wt);
    k_xsum<<<dim3(NB, NBR), 256, 0, stream>>>(x0, x1, x2, m0, m1, m2, xsum, nvalid);
    k_uv<<<dim3(NB, NBR), 256, 0, stream>>>(xsum, nvalid, W_attn, b_attn, u, cb);
    k_attn<<<dim3(NB, NBR), 256, 0, stream>>>(x0, x1, x2, m0, m1, m2, u, cb, attn);
    k_gemm_pool<<<dim3((NB*TT)/128, DD/128, NBR), 256, 0, stream>>>(
        x0, x1, x2, Wt, b1, attn, pooledh);
    k_final<<<dim3(NB, NBR), 256, 0, stream>>>(pooledh, W2, b2, ln_g, ln_b,
                                               W_last, b_last, out);
}
